// LSTMScratch_44908178047584
// MI455X (gfx1250) — matrix operand order repaired and checked by reading, not compiled, not run on hardware
//
#include <hip/hip_runtime.h>

// Problem dims (from reference): T=4096, I=512, H=1024, O=256
#define T_ 4096
#define I_ 512
#define H_ 1024
#define O_ 256
#define NBLK 64   // persistent workgroups in the scan kernel (H/16)

typedef __attribute__((ext_vector_type(16))) __bf16 v16bf;
typedef __attribute__((ext_vector_type(8)))  float  v8f;

union FragU { v16bf v; uint4 q[2]; };

// ---- WMMA fragment loaders (contiguous k order: lane half h holds k=16h..16h+15, same in A and B) ----

// A-matrix 16x32 bf16, row-major source, leading dim `ld` (elements).
// lane L: row m=L%16; Kbase=(L<16?0:16); VGPR0-3 = K=kb..kb+7 (16B),
// VGPR4-7 = K=kb+8..kb+15 (16B).
__device__ __forceinline__ v16bf frag_a(const __bf16* src, int ld, int lane) {
  const int m    = lane & 15;
  const int koff = (lane < 16) ? 0 : 16;
  const __bf16* p = src + (size_t)m * ld + koff;
  FragU u;
  u.q[0] = *(const uint4*)(p);
  u.q[1] = *(const uint4*)(p + 8);
  return u.v;
}

// B-matrix 32x16 bf16 where B[k][n] = w[n][k] (w row-major, leading dim ld).
// lane L: col n=L%16; lanes 0-15 hold K=0..15, lanes 16-31 hold K=16..31
// (contiguous along w's row -> two b128 loads).
__device__ __forceinline__ v16bf frag_b_rows(const __bf16* w, int ld, int lane) {
  const int n    = lane & 15;
  const int koff = (lane < 16) ? 0 : 16;
  const __bf16* p = w + (size_t)n * ld + koff;
  FragU u;
  u.q[0] = *(const uint4*)(p);
  u.q[1] = *(const uint4*)(p + 8);
  return u.v;
}

// B-matrix 32x16 with every column equal to vector hseg[0..31] (matvec
// broadcast trick): lane needs hseg[koff..koff+15] only (ds_load_b128 x2).
__device__ __forceinline__ v16bf frag_b_bcast(const __bf16* hseg, int lane) {
  const int koff = (lane < 16) ? 0 : 16;
  FragU u;
  u.q[0] = *(const uint4*)(hseg + koff);
  u.q[1] = *(const uint4*)(hseg + koff + 8);
  return u.v;
}

__device__ __forceinline__ v8f wmma_bf16(v16bf a, v16bf b, v8f c) {
  return __builtin_amdgcn_wmma_f32_16x16x32_bf16(
      /*neg_a=*/false, a, /*neg_b=*/false, b,
      /*c_mod=*/(short)0, c, /*reuse_a=*/false, /*reuse_b=*/false);
}

__device__ __forceinline__ float sigmoidf_(float x) {
  return 1.0f / (1.0f + __expf(-x));
}

// ------------------------- init: counter + h0 row -------------------------
__global__ void k_init(unsigned* bar, __bf16* hsb) {
  const int tid = threadIdx.x;
  if (tid == 0) *bar = 0u;
  // zero hsb row 0 (1024 bf16 = 512 dwords)
  if (tid < 512) ((unsigned*)hsb)[tid] = 0u;
}

// --------------------------- fp32 -> bf16 pack ----------------------------
__global__ void k_cvt(const float* __restrict__ s, __bf16* __restrict__ d, int n) {
  int i = blockIdx.x * blockDim.x + threadIdx.x;
  if (i < n) d[i] = (__bf16)s[i];
}

// ------------------ gates GEMM: gx_g = x @ Wx_g^T + b_g -------------------
// grid: (T/16, H/16, 4 gates), block: 32 (one wave per 16x16 output tile)
__global__ void __launch_bounds__(32)
k_gates_gemm(const __bf16* __restrict__ xbf,
             const __bf16* wf, const __bf16* wi, const __bf16* wc, const __bf16* wo,
             const float* bf_, const float* bi_, const float* bc_, const float* bo_,
             float* gf, float* gi, float* gc, float* go) {
  const int lane = threadIdx.x;
  const int t0 = blockIdx.x * 16;
  const int j0 = blockIdx.y * 16;
  const int g  = blockIdx.z;
  const __bf16* w = (g == 0) ? wf : (g == 1) ? wi : (g == 2) ? wc : wo;
  const float*  b = (g == 0) ? bf_ : (g == 1) ? bi_ : (g == 2) ? bc_ : bo_;
  float*      out = (g == 0) ? gf : (g == 1) ? gi : (g == 2) ? gc : go;

  v8f acc = {};
#pragma unroll 4
  for (int kc = 0; kc < I_ / 32; kc++) {
    v16bf a  = frag_a(xbf + (size_t)t0 * I_ + kc * 32, I_, lane);
    v16bf bb = frag_b_rows(w + (size_t)j0 * I_ + kc * 32, I_, lane);
    acc = wmma_bf16(a, bb, acc);
  }
  const int n  = lane & 15;
  const int mb = (lane < 16) ? 0 : 8;
  const float bias = b[j0 + n];
#pragma unroll
  for (int v = 0; v < 8; v++)
    out[(size_t)(t0 + mb + v) * H_ + j0 + n] = acc[v] + bias;
}

// ------------------------- persistent LSTM scan ---------------------------
// 64 WGs x 256 threads (8 waves). WG b owns h rows [16b, 16b+16).
// Wave w: gate g=w/2, K-half kh=w&1. Its 16x512 Wh slice lives in 128 VGPRs
// (16 bf16 A-fragments) for all 4096 steps. Per step: 16 WMMAs per wave
// (two independent accumulator chains), h_{t-1} staged to LDS with the
// CDNA5 async-to-LDS path, device-wide atomic barrier between steps.
__global__ void __launch_bounds__(256)
k_lstm_scan(const float* __restrict__ gxf, const float* __restrict__ gxi,
            const float* __restrict__ gxc, const float* __restrict__ gxo,
            const __bf16* __restrict__ whf, const __bf16* __restrict__ whi,
            const __bf16* __restrict__ whc, const __bf16* __restrict__ who,
            __bf16* __restrict__ hsb,   // (T_+1) x H_, row 0 = h0 = 0
            unsigned* __restrict__ bar) {
  __shared__ __bf16 hsh[H_];
  __shared__ float  part[8][16];

  const int tid  = threadIdx.x;
  const int lane = tid & 31;
  const int wv   = tid >> 5;
  const int gate = wv >> 1;
  const int kh   = wv & 1;
  const int kbase = kh * (H_ / 2);
  const int j0    = blockIdx.x * 16;

  const __bf16* wh =
      (gate == 0) ? whf : (gate == 1) ? whi : (gate == 2) ? whc : who;

  // Persistent weight fragments: Wh_g[j0..j0+15][kbase..kbase+511]
  v16bf afrag[16];
#pragma unroll
  for (int kc = 0; kc < 16; kc++)
    afrag[kc] = frag_a(wh + (size_t)j0 * H_ + kbase + kc * 32, H_, lane);

  float creg = 0.0f;  // cell state for j = j0 + tid (threads 0..15)

  // LDS byte address for this thread's async-staging destination.
  // (low 32 bits of a generic pointer to LDS == byte offset in LDS space)
  const unsigned lds_dst = (unsigned)(size_t)(&hsh[0]) + (unsigned)tid * 8u;
  const unsigned voff    = (unsigned)tid * 8u;

  for (int t = 0; t < T_; t++) {
    // stage h_{t-1} (bf16, 2KB) into LDS: 256 lanes x async 8B, no VGPR
    // round-trip (GLOBAL_LOAD_ASYNC_TO_LDS_B64, GVS mode, ASYNCcnt).
    {
      const unsigned long long sbase =
          (unsigned long long)(const void*)(hsb + (size_t)t * H_);
      asm volatile(
          "global_load_async_to_lds_b64 %0, %1, %2\n\t"
          "s_wait_asynccnt 0"
          :
          : "v"(lds_dst), "v"(voff), "s"(sbase)
          : "memory");
    }
    __syncthreads();

    // partial matvec: 16 x WMMA over this wave's K-half, two independent
    // accumulator chains to halve the WMMA RAW dependency depth.
    v8f acc0 = {}, acc1 = {};
#pragma unroll
    for (int kc = 0; kc < 16; kc += 2) {
      v16bf b0 = frag_b_bcast(hsh + kbase + kc * 32, lane);
      v16bf b1 = frag_b_bcast(hsh + kbase + (kc + 1) * 32, lane);
      acc0 = wmma_bf16(afrag[kc], b0, acc0);
      acc1 = wmma_bf16(afrag[kc + 1], b1, acc1);
    }

    // D columns are identical; lane0 holds m=0..7, lane16 holds m=8..15
    if (lane == 0 || lane == 16) {
      const int mb = (lane == 0) ? 0 : 8;
#pragma unroll
      for (int v = 0; v < 8; v++) part[wv][mb + v] = acc0[v] + acc1[v];
    }
    __syncthreads();

    // gate epilogue on 16 lanes (j = j0 + tid)
    if (tid < 16) {
      const size_t row = (size_t)t * H_ + (j0 + tid);
      float gf = gxf[row] + part[0][tid] + part[1][tid];
      float gi = gxi[row] + part[2][tid] + part[3][tid];
      float gc = gxc[row] + part[4][tid] + part[5][tid];
      float go = gxo[row] + part[6][tid] + part[7][tid];
      float f_t = sigmoidf_(gf);
      float i_t = sigmoidf_(gi);
      float cd  = tanhf(gc);
      creg = f_t * creg + i_t * cd;
      float o_t = sigmoidf_(go);
      float h_t = o_t * creg;            // reference omits tanh(c_t)
      hsb[(size_t)(t + 1) * H_ + (j0 + tid)] = (__bf16)h_t;

      // overlap next step's gx fetch with the grid barrier
      if (t + 1 < T_) {
        const size_t nrow = row + H_;
        __builtin_prefetch(&gxf[nrow], 0, 1);
        __builtin_prefetch(&gxi[nrow], 0, 1);
        __builtin_prefetch(&gxc[nrow], 0, 1);
        __builtin_prefetch(&gxo[nrow], 0, 1);
      }
    }

    // device-wide barrier (release h_t, acquire everyone else's h_t)
    __threadfence();
    __syncthreads();
    if (tid == 0) {
      atomicAdd(bar, 1u);
      const unsigned target = (unsigned)(NBLK * (t + 1));
      while (__hip_atomic_load(bar, __ATOMIC_ACQUIRE,
                               __HIP_MEMORY_SCOPE_AGENT) < target)
        __builtin_amdgcn_s_sleep(2);
    }
    __syncthreads();
    __threadfence();
  }
}

// --------------------- output GEMM: y = hs @ Wy^T + by --------------------
// grid: (T/16, O/16), block 32
__global__ void __launch_bounds__(32)
k_out_gemm(const __bf16* __restrict__ hsb,  // rows 1..T of (T+1) x H
           const __bf16* __restrict__ wy, const float* __restrict__ by,
           float* __restrict__ y) {
  const int lane = threadIdx.x;
  const int t0 = blockIdx.x * 16;
  const int o0 = blockIdx.y * 16;
  const __bf16* hs = hsb + H_;  // skip h0 row

  v8f acc = {};
#pragma unroll 4
  for (int kc = 0; kc < H_ / 32; kc++) {
    v16bf a  = frag_a(hs + (size_t)t0 * H_ + kc * 32, H_, lane);
    v16bf bb = frag_b_rows(wy + (size_t)o0 * H_ + kc * 32, H_, lane);
    acc = wmma_bf16(a, bb, acc);
  }
  const int n  = lane & 15;
  const int mb = (lane < 16) ? 0 : 8;
  const float bias = by[o0 + n];
#pragma unroll
  for (int v = 0; v < 8; v++)
    y[(size_t)(t0 + mb + v) * O_ + o0 + n] = acc[v] + bias;
}

// ------------------------------- launcher ---------------------------------
static inline size_t align256(size_t x) { return (x + 255) & ~(size_t)255; }

extern "C" void kernel_launch(void* const* d_in, const int* in_sizes, int n_in,
                              void* d_out, int out_size, void* d_ws, size_t ws_size,
                              hipStream_t stream) {
  (void)in_sizes; (void)n_in; (void)out_size; (void)ws_size;

  const float* x   = (const float*)d_in[0];
  const float* Wxf = (const float*)d_in[1];
  const float* Whf = (const float*)d_in[2];
  const float* bfv = (const float*)d_in[3];
  const float* Wxi = (const float*)d_in[4];
  const float* Whi = (const float*)d_in[5];
  const float* biv = (const float*)d_in[6];
  const float* Wxc = (const float*)d_in[7];
  const float* Whc = (const float*)d_in[8];
  const float* bcv = (const float*)d_in[9];
  const float* Wxo = (const float*)d_in[10];
  const float* Who = (const float*)d_in[11];
  const float* bov = (const float*)d_in[12];
  const float* Wy  = (const float*)d_in[13];
  const float* by  = (const float*)d_in[14];
  float* y = (float*)d_out;

  // workspace carve-up (~93 MB)
  char* ws = (char*)d_ws;
  size_t off = 0;
  unsigned* bar = (unsigned*)(ws + off); off += align256(sizeof(unsigned));
  __bf16* xbf   = (__bf16*)(ws + off);   off += align256((size_t)T_ * I_ * 2);
  __bf16* wxb[4]; for (int g = 0; g < 4; g++) { wxb[g] = (__bf16*)(ws + off); off += align256((size_t)H_ * I_ * 2); }
  __bf16* whb[4]; for (int g = 0; g < 4; g++) { whb[g] = (__bf16*)(ws + off); off += align256((size_t)H_ * H_ * 2); }
  __bf16* wyb   = (__bf16*)(ws + off);   off += align256((size_t)O_ * H_ * 2);
  __bf16* hsb   = (__bf16*)(ws + off);   off += align256((size_t)(T_ + 1) * H_ * 2);
  float* gx[4];   for (int g = 0; g < 4; g++) { gx[g] = (float*)(ws + off); off += align256((size_t)T_ * H_ * 4); }

  // 0) init barrier counter and h0 row
  k_init<<<1, 512, 0, stream>>>(bar, hsb);

  // 1) fp32 -> bf16 packs
  auto cvt = [&](const float* s, __bf16* d, int n) {
    k_cvt<<<(n + 255) / 256, 256, 0, stream>>>(s, d, n);
  };
  cvt(x, xbf, T_ * I_);
  cvt(Wxf, wxb[0], H_ * I_); cvt(Wxi, wxb[1], H_ * I_);
  cvt(Wxc, wxb[2], H_ * I_); cvt(Wxo, wxb[3], H_ * I_);
  cvt(Whf, whb[0], H_ * H_); cvt(Whi, whb[1], H_ * H_);
  cvt(Whc, whb[2], H_ * H_); cvt(Who, whb[3], H_ * H_);
  cvt(Wy, wyb, O_ * H_);

  // 2) input projections for all timesteps (WMMA GEMM)
  dim3 gg(T_ / 16, H_ / 16, 4);
  k_gates_gemm<<<gg, 32, 0, stream>>>(xbf,
      wxb[0], wxb[1], wxb[2], wxb[3],
      bfv, biv, bcv, bov,
      gx[0], gx[1], gx[2], gx[3]);

  // 3) persistent recurrent scan (weights register-resident, grid-synced)
  k_lstm_scan<<<NBLK, 256, 0, stream>>>(
      gx[0], gx[1], gx[2], gx[3],
      whb[0], whb[1], whb[2], whb[3],
      hsb, bar);

  // 4) output projection
  dim3 go(T_ / 16, O_ / 16);
  k_out_gemm<<<go, 32, 0, stream>>>(hsb, wyb, by, y);
}
